// SwinTransformerBlock_37469294690494
// MI455X (gfx1250) — compile-verified, matrix operand order checked by reading, not run on hardware
//
#include <hip/hip_runtime.h>
#include <hip/hip_bf16.h>

// ---------------- types ----------------
typedef __attribute__((ext_vector_type(16))) __bf16 v16bf;
typedef __attribute__((ext_vector_type(8)))  float  v8f;
typedef int v4i __attribute__((vector_size(4 * sizeof(int))));

struct alignas(16) Chunk { unsigned int a, b, c, d; };

union Frag {
    Chunk          ch[2];
    unsigned short u[16];
    v16bf          v;
};

__device__ __forceinline__ unsigned short f32_to_bf16(float f) {
    unsigned int u = __float_as_uint(f);
    unsigned int r = u + 0x7FFFu + ((u >> 16) & 1u);
    return (unsigned short)(r >> 16);
}
__device__ __forceinline__ float bf16_to_f32(unsigned short h) {
    return __uint_as_float(((unsigned int)h) << 16);
}
__device__ __forceinline__ v8f v8f_zero() {
    v8f z;
#pragma unroll
    for (int i = 0; i < 8; ++i) z[i] = 0.0f;
    return z;
}

#define WMMA_BF16(A, B, C) \
    __builtin_amdgcn_wmma_f32_16x16x32_bf16(false, (A), false, (B), (short)0, (C), false, false)

// ---------------- async global -> LDS copy (16 bytes), ASYNCcnt-tracked ----------------
#if __has_builtin(__builtin_amdgcn_global_load_async_to_lds_b128)
#define HAVE_ASYNC_LDS 1
#else
#define HAVE_ASYNC_LDS 0
#endif

typedef v4i __attribute__((address_space(1)))* gptr4_t;
typedef v4i __attribute__((address_space(3)))* lptr4_t;

__device__ __forceinline__ void async_cp16(const unsigned short* g, unsigned short* l) {
#if HAVE_ASYNC_LDS
    __builtin_amdgcn_global_load_async_to_lds_b128(
        (gptr4_t)(unsigned long long)(uintptr_t)g,  // global VA == flat VA
        (lptr4_t)(void*)l,                          // addrspacecast generic -> LDS
        0, 0);
#else
    *(Chunk*)l = *(const Chunk*)g;
#endif
}

#if HAVE_ASYNC_LDS
#if __has_builtin(__builtin_amdgcn_s_wait_asynccnt)
#define WAIT_ASYNC(n) __builtin_amdgcn_s_wait_asynccnt(n)
#else
#define WAIT_ASYNC(n) asm volatile("s_wait_asynccnt %0" ::"i"(n) : "memory")
#endif
#else
#define WAIT_ASYNC(n) ((void)0)
#endif

// ---------------- fp32 -> bf16 convert ----------------
__global__ void cvt_bf16_kernel(const float* __restrict__ in, unsigned short* __restrict__ out) {
    long i = (long)blockIdx.x * 256 + threadIdx.x;
    out[i] = f32_to_bf16(in[i]);
}

// ---------------- CPB relative-position bias ----------------
// bias[h][n][m], h<12, n,m < 64   (one block, 256 threads)
__global__ void cpb_bias_kernel(const float* __restrict__ w1,   // (512,2)
                                const float* __restrict__ b1,   // (512)
                                const float* __restrict__ w2,   // (12,512)
                                float* __restrict__ bias) {
    __shared__ float table[225 * 12];
    const int tid = threadIdx.x;
    if (tid < 225) {
        float dy = (float)(tid / 15 - 7);
        float dx = (float)(tid % 15 - 7);
        float t0 = dy / 7.0f * 8.0f;
        float t1 = dx / 7.0f * 8.0f;
        float c0 = copysignf(log2f(fabsf(t0) + 1.0f) / 3.0f, t0);
        float c1 = copysignf(log2f(fabsf(t1) + 1.0f) / 3.0f, t1);
        float acc[12];
#pragma unroll
        for (int h = 0; h < 12; ++h) acc[h] = 0.0f;
        for (int j = 0; j < 512; ++j) {
            float hj = fmaxf(w1[2 * j] * c0 + w1[2 * j + 1] * c1 + b1[j], 0.0f);
#pragma unroll
            for (int h = 0; h < 12; ++h) acc[h] += w2[h * 512 + j] * hj;
        }
#pragma unroll
        for (int h = 0; h < 12; ++h) table[tid * 12 + h] = acc[h];
    }
    __syncthreads();
    for (int idx = tid; idx < 12 * 64 * 64; idx += 256) {
        int h = idx >> 12;
        int n = (idx >> 6) & 63;
        int m = idx & 63;
        int rel = ((n >> 3) - (m >> 3) + 7) * 15 + ((n & 7) - (m & 7) + 7);
        bias[idx] = table[rel * 12 + h];
    }
}

// ---------------- LayerNorm -> bf16 (one wave per token, C=384) ----------------
__global__ __launch_bounds__(256) void ln_bf16_kernel(const float* __restrict__ x,
                                                      const float* __restrict__ g,
                                                      const float* __restrict__ b,
                                                      unsigned short* __restrict__ out) {
    const int lane = threadIdx.x & 31;
    const int wave = threadIdx.x >> 5;
    const long token = (long)blockIdx.x * 8 + wave;
    const float* row = x + token * 384;
    float v[12];
    float s = 0.0f;
#pragma unroll
    for (int i = 0; i < 12; ++i) { v[i] = row[lane + i * 32]; s += v[i]; }
#pragma unroll
    for (int m = 16; m > 0; m >>= 1) s += __shfl_xor(s, m, 32);
    float mean = s * (1.0f / 384.0f);
    float vs = 0.0f;
#pragma unroll
    for (int i = 0; i < 12; ++i) { float d = v[i] - mean; vs += d * d; }
#pragma unroll
    for (int m = 16; m > 0; m >>= 1) vs += __shfl_xor(vs, m, 32);
    float rstd = rsqrtf(vs * (1.0f / 384.0f) + 1e-5f);
#pragma unroll
    for (int i = 0; i < 12; ++i) {
        int idx = lane + i * 32;
        float o = (v[i] - mean) * rstd * g[idx] + b[idx];
        out[token * 384 + idx] = f32_to_bf16(o);
    }
}

// ---------------- Generic bf16 GEMM:  out[M][N] = A[M][K] * B[N][K]^T + bias ----------------
// Double-buffered LDS, async global->LDS staging (4 async ops / wave / tile).
// MODE 0: bf16 out;  MODE 1: fp32 out with fp32 residual added
template <int MODE>
__global__ __launch_bounds__(256) void gemm_bf16_kernel(const unsigned short* __restrict__ A,
                                                        const unsigned short* __restrict__ B,
                                                        const float* __restrict__ bias,
                                                        const float* __restrict__ res,
                                                        void* __restrict__ out,
                                                        int M, int N, int K) {
    __shared__ __align__(16) unsigned short As[2][128 * 32];
    __shared__ __align__(16) unsigned short Bs[2][128 * 32];
    const int tid = threadIdx.x;
    const int lane = tid & 31;
    const int wave = tid >> 5;
    const int wm = wave & 3, wn = wave >> 2;
    const int half = lane >> 4, l16 = lane & 15;
    const long bm = (long)blockIdx.x * 128;
    const long bn = (long)blockIdx.y * 128;

    const int row0 = tid >> 2;          // 0..63
    const int row1 = row0 + 64;         // 64..127
    const int col0 = (tid & 3) * 8;

    v8f acc[2][4];
#pragma unroll
    for (int i = 0; i < 2; ++i)
#pragma unroll
        for (int j = 0; j < 4; ++j) acc[i][j] = v8f_zero();

    auto issue = [&](int buf, int k0) {
        async_cp16(A + (bm + row0) * (long)K + k0 + col0, &As[buf][row0 * 32 + col0]);
        async_cp16(A + (bm + row1) * (long)K + k0 + col0, &As[buf][row1 * 32 + col0]);
        async_cp16(B + (bn + row0) * (long)K + k0 + col0, &Bs[buf][row0 * 32 + col0]);
        async_cp16(B + (bn + row1) * (long)K + k0 + col0, &Bs[buf][row1 * 32 + col0]);
    };

    issue(0, 0);
    const int nk = K >> 5;
    for (int it = 0; it < nk; ++it) {
        const int b = it & 1;
        const int k0 = it << 5;
        if (it + 1 < nk) {
            issue(b ^ 1, k0 + 32);   // prefetch next tile while computing this one
            WAIT_ASYNC(4);           // first 4 (this tile) complete, in-order
        } else {
            WAIT_ASYNC(0);
        }
        __syncthreads();
        v16bf af[2], bfv[4];
#pragma unroll
        for (int mt = 0; mt < 2; ++mt) {
            int row = wm * 32 + mt * 16 + l16;
            Frag f;
            f.ch[0] = *(const Chunk*)&As[b][row * 32 + 8 * half];
            f.ch[1] = *(const Chunk*)&As[b][row * 32 + 16 + 8 * half];
            af[mt] = f.v;
        }
#pragma unroll
        for (int nt = 0; nt < 4; ++nt) {
            int row = wn * 64 + nt * 16 + l16;
            Frag f;
            f.ch[0] = *(const Chunk*)&Bs[b][row * 32 + 8 * half];
            f.ch[1] = *(const Chunk*)&Bs[b][row * 32 + 16 + 8 * half];
            bfv[nt] = f.v;
        }
#pragma unroll
        for (int mt = 0; mt < 2; ++mt)
#pragma unroll
            for (int nt = 0; nt < 4; ++nt) acc[mt][nt] = WMMA_BF16(af[mt], bfv[nt], acc[mt][nt]);
        __syncthreads();
    }

#pragma unroll
    for (int mt = 0; mt < 2; ++mt)
#pragma unroll
        for (int nt = 0; nt < 4; ++nt)
#pragma unroll
            for (int r = 0; r < 8; ++r) {
                long gm = bm + wm * 32 + mt * 16 + r + 8 * half;
                long gn = bn + wn * 64 + nt * 16 + l16;
                float v = acc[mt][nt][r] + bias[gn];
                if (MODE == 1) {
                    v += res[gm * N + gn];
                    ((float*)out)[gm * N + gn] = v;
                } else {
                    ((unsigned short*)out)[gm * N + gn] = f32_to_bf16(v);
                }
            }
}

// ---------------- W12 dual GEMM with fused SwiGLU ----------------
// out[m][j] = silu(A*B[j+1536] + b[j+1536]) * (A*B[j] + b[j]),  j < 1536, bf16 out
__global__ __launch_bounds__(256) void gemm_w12_swiglu_kernel(const unsigned short* __restrict__ A,
                                                              const unsigned short* __restrict__ B,
                                                              const float* __restrict__ bias,
                                                              unsigned short* __restrict__ out,
                                                              int K) {
    __shared__ __align__(16) unsigned short As[2][128 * 32];
    __shared__ __align__(16) unsigned short B1s[2][64 * 32];
    __shared__ __align__(16) unsigned short B2s[2][64 * 32];
    const int tid = threadIdx.x;
    const int lane = tid & 31;
    const int wave = tid >> 5;
    const int wm = wave & 3, wn = wave >> 2;
    const int half = lane >> 4, l16 = lane & 15;
    const long bm = (long)blockIdx.x * 128;
    const long bn = (long)blockIdx.y * 64;

    const int row0 = tid >> 2;          // 0..63
    const int row1 = row0 + 64;
    const int col0 = (tid & 3) * 8;

    v8f a1[2][2], a2[2][2];
#pragma unroll
    for (int i = 0; i < 2; ++i)
#pragma unroll
        for (int j = 0; j < 2; ++j) { a1[i][j] = v8f_zero(); a2[i][j] = v8f_zero(); }

    auto issue = [&](int buf, int k0) {
        async_cp16(A + (bm + row0) * (long)K + k0 + col0, &As[buf][row0 * 32 + col0]);
        async_cp16(A + (bm + row1) * (long)K + k0 + col0, &As[buf][row1 * 32 + col0]);
        async_cp16(B + (bn + row0) * (long)K + k0 + col0, &B1s[buf][row0 * 32 + col0]);
        async_cp16(B + (bn + 1536 + row0) * (long)K + k0 + col0, &B2s[buf][row0 * 32 + col0]);
    };

    issue(0, 0);
    const int nk = K >> 5;
    for (int it = 0; it < nk; ++it) {
        const int b = it & 1;
        const int k0 = it << 5;
        if (it + 1 < nk) {
            issue(b ^ 1, k0 + 32);
            WAIT_ASYNC(4);
        } else {
            WAIT_ASYNC(0);
        }
        __syncthreads();
        v16bf af[2], b1f[2], b2f[2];
#pragma unroll
        for (int mt = 0; mt < 2; ++mt) {
            int row = wm * 32 + mt * 16 + l16;
            Frag f;
            f.ch[0] = *(const Chunk*)&As[b][row * 32 + 8 * half];
            f.ch[1] = *(const Chunk*)&As[b][row * 32 + 16 + 8 * half];
            af[mt] = f.v;
        }
#pragma unroll
        for (int nt = 0; nt < 2; ++nt) {
            int row = wn * 32 + nt * 16 + l16;
            Frag f;
            f.ch[0] = *(const Chunk*)&B1s[b][row * 32 + 8 * half];
            f.ch[1] = *(const Chunk*)&B1s[b][row * 32 + 16 + 8 * half];
            b1f[nt] = f.v;
            Frag g;
            g.ch[0] = *(const Chunk*)&B2s[b][row * 32 + 8 * half];
            g.ch[1] = *(const Chunk*)&B2s[b][row * 32 + 16 + 8 * half];
            b2f[nt] = g.v;
        }
#pragma unroll
        for (int mt = 0; mt < 2; ++mt)
#pragma unroll
            for (int nt = 0; nt < 2; ++nt) {
                a1[mt][nt] = WMMA_BF16(af[mt], b1f[nt], a1[mt][nt]);
                a2[mt][nt] = WMMA_BF16(af[mt], b2f[nt], a2[mt][nt]);
            }
        __syncthreads();
    }

#pragma unroll
    for (int mt = 0; mt < 2; ++mt)
#pragma unroll
        for (int nt = 0; nt < 2; ++nt)
#pragma unroll
            for (int r = 0; r < 8; ++r) {
                long gm = bm + wm * 32 + mt * 16 + r + 8 * half;
                long j = bn + wn * 32 + nt * 16 + l16;
                float v1 = a1[mt][nt][r] + bias[j];
                float v2 = a2[mt][nt][r] + bias[j + 1536];
                float sw = v2 / (1.0f + __expf(-v2));  // silu(x2)
                out[gm * 1536 + j] = f32_to_bf16(sw * v1);
            }
}

// ---------------- Windowed cosine attention ----------------
// grid (2048 windows, 12 heads), 64 threads (2 waves). qkv bf16 [tok][1152], out bf16 [tok][384]
__global__ __launch_bounds__(64) void attn_kernel(const unsigned short* __restrict__ qkv,
                                                  const float* __restrict__ bias,
                                                  const float* __restrict__ logit_scale,
                                                  unsigned short* __restrict__ out) {
    __shared__ __align__(16) unsigned short qs[64 * 32];
    __shared__ __align__(16) unsigned short ks[64 * 32];
    __shared__ __align__(16) unsigned short vts[32 * 64];  // transposed V: vts[d][m]
    __shared__ int toks[64];
    __shared__ int regi[64];

    const int wid = blockIdx.x;
    const int h = blockIdx.y;
    const int n = threadIdx.x;
    const int bimg = wid >> 6;
    const int wy = (wid >> 3) & 7;
    const int wx = wid & 7;
    const int rr0 = n >> 3, cc0 = n & 7;
    const int sh = wy * 8 + rr0, sw = wx * 8 + cc0;
    const int oh = (sh + 4) & 63, ow = (sw + 4) & 63;
    const long tok = (long)bimg * 4096 + oh * 64 + ow;
    toks[n] = (int)tok;
    int rh = sh < 56 ? 0 : (sh < 60 ? 1 : 2);
    int rw = sw < 56 ? 0 : (sw < 60 ? 1 : 2);
    regi[n] = rh * 3 + rw;

    const float ls = __expf(fminf(logit_scale[h], 4.6051702f));  // ln(100)

    // stage q (normalized*ls), k (normalized), v^T as bf16
    {
        const unsigned short* qrow = qkv + tok * 1152 + h * 32;
        float qf[32], kf[32], vf[32];
        float qss = 0.0f, kss = 0.0f;
#pragma unroll
        for (int d = 0; d < 32; ++d) {
            qf[d] = bf16_to_f32(qrow[d]);
            kf[d] = bf16_to_f32(qrow[384 + d]);
            vf[d] = bf16_to_f32(qrow[768 + d]);
            qss += qf[d] * qf[d];
            kss += kf[d] * kf[d];
        }
        float qsc = ls / fmaxf(sqrtf(qss), 1e-12f);
        float ksc = 1.0f / fmaxf(sqrtf(kss), 1e-12f);
#pragma unroll
        for (int d = 0; d < 32; ++d) {
            qs[n * 32 + d] = f32_to_bf16(qf[d] * qsc);
            ks[n * 32 + d] = f32_to_bf16(kf[d] * ksc);
            vts[d * 64 + n] = f32_to_bf16(vf[d]);
        }
    }
    __syncthreads();

    const int lane = threadIdx.x & 31;
    const int w = threadIdx.x >> 5;   // wave id: handles query tiles {2w, 2w+1}
    const int half = lane >> 4, l16 = lane & 15;

    // S^T = K * Q^T  (M = key index on regs, N = query index on lanes)
    v16bf kfr[4], qfr[2];
#pragma unroll
    for (int mt = 0; mt < 4; ++mt) {
        int row = mt * 16 + l16;
        Frag f;
        f.ch[0] = *(const Chunk*)&ks[row * 32 + 8 * half];
        f.ch[1] = *(const Chunk*)&ks[row * 32 + 16 + 8 * half];
        kfr[mt] = f.v;
    }
#pragma unroll
    for (int nt = 0; nt < 2; ++nt) {
        int row = w * 32 + nt * 16 + l16;
        Frag f;
        f.ch[0] = *(const Chunk*)&qs[row * 32 + 8 * half];
        f.ch[1] = *(const Chunk*)&qs[row * 32 + 16 + 8 * half];
        qfr[nt] = f.v;
    }

    v8f s[4][2];
#pragma unroll
    for (int mt = 0; mt < 4; ++mt)
#pragma unroll
        for (int nt = 0; nt < 2; ++nt) s[mt][nt] = WMMA_BF16(kfr[mt], qfr[nt], v8f_zero());

    // + relative position bias + arithmetic shift-mask, then softmax over keys
#pragma unroll
    for (int nt = 0; nt < 2; ++nt) {
        int qidx = w * 32 + nt * 16 + l16;
        int qreg = regi[qidx];
        const float* brow = bias + ((long)h * 64 + qidx) * 64;
#pragma unroll
        for (int mt = 0; mt < 4; ++mt)
#pragma unroll
            for (int r = 0; r < 8; ++r) {
                int mk = mt * 16 + r + 8 * half;
                s[mt][nt][r] += brow[mk] + (regi[mk] == qreg ? 0.0f : -1e30f);
            }
        float mx = -1e30f;
#pragma unroll
        for (int mt = 0; mt < 4; ++mt)
#pragma unroll
            for (int r = 0; r < 8; ++r) mx = fmaxf(mx, s[mt][nt][r]);
        mx = fmaxf(mx, __shfl_xor(mx, 16, 32));
        float sum = 0.0f;
#pragma unroll
        for (int mt = 0; mt < 4; ++mt)
#pragma unroll
            for (int r = 0; r < 8; ++r) {
                float e = __expf(s[mt][nt][r] - mx);
                s[mt][nt][r] = e;
                sum += e;
            }
        sum += __shfl_xor(sum, 16, 32);
        float inv = 1.0f / sum;
#pragma unroll
        for (int mt = 0; mt < 4; ++mt)
#pragma unroll
            for (int r = 0; r < 8; ++r) s[mt][nt][r] *= inv;
    }

    // C-layout of S^T maps directly into A-operand layout for P@V (no shuffles)
    v16bf a2[2][2];
#pragma unroll
    for (int nt = 0; nt < 2; ++nt)
#pragma unroll
        for (int kt = 0; kt < 2; ++kt) {
            Frag f;
#pragma unroll
            for (int j = 0; j < 8; ++j) {
                f.u[j] = f32_to_bf16(s[2 * kt][nt][j]);
                f.u[8 + j] = f32_to_bf16(s[2 * kt + 1][nt][j]);
            }
            a2[nt][kt] = f.v;
        }
    v16bf b2[2][2];
#pragma unroll
    for (int dt = 0; dt < 2; ++dt)
#pragma unroll
        for (int kt = 0; kt < 2; ++kt) {
            int row = dt * 16 + l16;
            Frag f;
            f.ch[0] = *(const Chunk*)&vts[row * 64 + kt * 32 + 8 * half];
            f.ch[1] = *(const Chunk*)&vts[row * 64 + kt * 32 + 16 + 8 * half];
            b2[dt][kt] = f.v;
        }

#pragma unroll
    for (int nt = 0; nt < 2; ++nt)
#pragma unroll
        for (int dt = 0; dt < 2; ++dt) {
            v8f o = WMMA_BF16(a2[nt][0], b2[dt][0], v8f_zero());
            o = WMMA_BF16(a2[nt][1], b2[dt][1], o);
#pragma unroll
            for (int r = 0; r < 8; ++r) {
                int qidx = w * 32 + nt * 16 + r + 8 * half;
                int d = dt * 16 + l16;
                long t = (long)toks[qidx];
                out[t * 384 + h * 32 + d] = f32_to_bf16(o[r]);
            }
        }
}

// ---------------- host launcher ----------------
extern "C" void kernel_launch(void* const* d_in, const int* in_sizes, int n_in,
                              void* d_out, int out_size, void* d_ws, size_t ws_size,
                              hipStream_t stream) {
    (void)in_sizes; (void)n_in; (void)out_size; (void)ws_size;
    const long M = 131072;  // 32 * 4096 tokens

    const float* x        = (const float*)d_in[0];
    const float* norm1_g  = (const float*)d_in[3];
    const float* norm1_b  = (const float*)d_in[4];
    const float* qkv_w    = (const float*)d_in[5];
    const float* qkv_b    = (const float*)d_in[6];
    const float* proj_w   = (const float*)d_in[7];
    const float* proj_b   = (const float*)d_in[8];
    const float* cpb_w1   = (const float*)d_in[9];
    const float* cpb_b1   = (const float*)d_in[10];
    const float* cpb_w2   = (const float*)d_in[11];
    const float* lscale   = (const float*)d_in[12];
    const float* norm2_g  = (const float*)d_in[13];
    const float* norm2_b  = (const float*)d_in[14];
    const float* w12_w    = (const float*)d_in[15];
    const float* w12_b    = (const float*)d_in[16];
    const float* w3_w     = (const float*)d_in[17];
    const float* w3_b     = (const float*)d_in[18];

    // workspace layout
    char* base = (char*)d_ws;
    size_t off = 0;
    auto alloc = [&](size_t bytes) {
        off = (off + 255) & ~(size_t)255;
        size_t o = off;
        off += bytes;
        return o;
    };
    unsigned short* qkvw_bf = (unsigned short*)(base + alloc(1152 * 384 * 2));
    unsigned short* projw_bf = (unsigned short*)(base + alloc(384 * 384 * 2));
    unsigned short* w12w_bf  = (unsigned short*)(base + alloc(3072 * 384 * 2));
    unsigned short* w3w_bf   = (unsigned short*)(base + alloc(384 * 1536 * 2));
    float*          biasbuf  = (float*)(base + alloc(12 * 64 * 64 * 4));
    unsigned short* xn       = (unsigned short*)(base + alloc((size_t)M * 384 * 2));
    unsigned short* big      = (unsigned short*)(base + alloc((size_t)M * 1536 * 2)); // qkv then ffn
    unsigned short* attn     = (unsigned short*)(base + alloc((size_t)M * 384 * 2));
    float*          x2       = (float*)(base + alloc((size_t)M * 384 * 4));

    // 1. weights -> bf16
    cvt_bf16_kernel<<<1152 * 384 / 256, 256, 0, stream>>>(qkv_w, qkvw_bf);
    cvt_bf16_kernel<<<384 * 384 / 256, 256, 0, stream>>>(proj_w, projw_bf);
    cvt_bf16_kernel<<<3072 * 384 / 256, 256, 0, stream>>>(w12_w, w12w_bf);
    cvt_bf16_kernel<<<384 * 1536 / 256, 256, 0, stream>>>(w3_w, w3w_bf);

    // 2. CPB bias table
    cpb_bias_kernel<<<1, 256, 0, stream>>>(cpb_w1, cpb_b1, cpb_w2, biasbuf);

    // 3. LN1 -> bf16
    ln_bf16_kernel<<<M / 8, 256, 0, stream>>>(x, norm1_g, norm1_b, xn);

    // 4. QKV GEMM (M x 384 -> M x 1152, bf16 out)
    gemm_bf16_kernel<0><<<dim3(M / 128, 1152 / 128), 256, 0, stream>>>(
        xn, qkvw_bf, qkv_b, nullptr, big, (int)M, 1152, 384);

    // 5. windowed cosine attention
    attn_kernel<<<dim3(2048, 12), 64, 0, stream>>>(big, biasbuf, lscale, attn);

    // 6. proj GEMM + residual(x) -> x2 (fp32)
    gemm_bf16_kernel<1><<<dim3(M / 128, 384 / 128), 256, 0, stream>>>(
        attn, projw_bf, proj_b, x, x2, (int)M, 384, 384);

    // 7. LN2 -> bf16
    ln_bf16_kernel<<<M / 8, 256, 0, stream>>>(x2, norm2_g, norm2_b, xn);

    // 8. W12 dual GEMM + fused SwiGLU -> ffn (bf16, M x 1536)
    gemm_w12_swiglu_kernel<<<dim3(M / 128, 1536 / 64), 256, 0, stream>>>(
        xn, w12w_bf, w12_b, big, 384);

    // 9. W3 GEMM + residual(x2) -> d_out (fp32)
    gemm_bf16_kernel<1><<<dim3(M / 128, 384 / 128), 256, 0, stream>>>(
        big, w3w_bf, w3_b, x2, d_out, (int)M, 384, 1536);
}
